// GlobalCAN_58729382805972
// MI455X (gfx1250) — matrix operand order repaired and checked by reading, not compiled, not run on hardware
//
#include <hip/hip_runtime.h>
#include <hip/hip_bf16.h>
#include <math.h>

#define S_DIM 512
#define B_DIM 1024
#define D_DIM 256
#define O_DIM 256
#define K_DIM 10

typedef __bf16    v16bf __attribute__((ext_vector_type(16)));
typedef float     v8f   __attribute__((ext_vector_type(8)));
typedef unsigned  v8u   __attribute__((ext_vector_type(8)));
typedef unsigned  v4u   __attribute__((ext_vector_type(4)));
typedef int       v8i   __attribute__((ext_vector_type(8)));
typedef int       v4i   __attribute__((ext_vector_type(4)));

union Frag {
    v16bf v;
    v8u   u8;
    v4u   u4[2];
    unsigned u[8];
};

__device__ __forceinline__ unsigned bf16_rne(float f) {
    unsigned u = __float_as_uint(f);
    return (u + 0x7fffu + ((u >> 16) & 1u)) >> 16;
}
__device__ __forceinline__ float bf16_to_f(unsigned short h) {
    return __uint_as_float(((unsigned)h) << 16);
}

// ---------------------------------------------------------------------------
// Repack w_t [D,O] fp32 -> bf16 WMMA B-fragments.
//   lanes 0-15:  K = 2j+h (+8 if j>=4)    (dword j, half h), N = lane&15
//   lanes 16-31: K = 8+2j+h (+8 if j>=4), N = lane&15
// Stored as [kk(8)][ntile(16)][lane(32)][j(8)] dwords -> 32B contiguous/lane.
// ---------------------------------------------------------------------------
__global__ __launch_bounds__(256) void pack_wt_kernel(const float* __restrict__ w_t,
                                                      unsigned* __restrict__ out) {
    int idx  = blockIdx.x * 256 + threadIdx.x;      // 32768 dwords total
    int j    = idx & 7;
    int lane = (idx >> 3) & 31;
    int nt   = (idx >> 8) & 15;
    int kk   = idx >> 12;
    int kbase = kk * 32 + ((lane & 16) ? 8 : 0) + 2 * j + ((j >= 4) ? 8 : 0);
    int n     = nt * 16 + (lane & 15);
    unsigned lo = bf16_rne(w_t[(size_t)kbase * O_DIM + n]);
    unsigned hi = bf16_rne(w_t[(size_t)(kbase + 1) * O_DIM + n]);
    out[idx] = lo | (hi << 16);
}

// ---------------------------------------------------------------------------
// Repack w_p [K,S,O] fp32 -> bf16 B-fragments for the sim GEMM.
// Reduction dim = O (256); N dim = k (10, zero-padded to 16).
// Stored as [s(512)][kk(8)][lane(32)][j(8)] dwords  (8 KB per s).
// ---------------------------------------------------------------------------
__global__ __launch_bounds__(256) void pack_wp_kernel(const float* __restrict__ w_p,
                                                      unsigned* __restrict__ out) {
    int idx  = blockIdx.x * 256 + threadIdx.x;      // 1,048,576 dwords total
    int j    = idx & 7;
    int lane = (idx >> 3) & 31;
    int kk   = (idx >> 8) & 7;
    int s    = idx >> 11;
    int n    = lane & 15;                           // k index
    int z    = kk * 32 + ((lane & 16) ? 8 : 0) + 2 * j + ((j >= 4) ? 8 : 0);   // o index
    unsigned lo = 0, hi = 0;
    if (n < K_DIM) {
        lo = bf16_rne(w_p[(size_t)n * S_DIM * O_DIM + (size_t)s * O_DIM + z]);
        hi = bf16_rne(w_p[(size_t)n * S_DIM * O_DIM + (size_t)s * O_DIM + z + 1]);
    }
    out[idx] = lo | (hi << 16);
}

// ---------------------------------------------------------------------------
// Main kernel: per block = one s, 128 rows of B.  8 waves, one 16-row strip
// each.  At entry wave 0 kicks a TDM tensor_load_to_lds of this s's w_p
// fragments (8 KB) which overlaps all of phase 1.
// Phase 1: session_emb strip = x_tile @ w_t via bf16 WMMA, masked, stored
// bf16; sigmoid into LDS.  Phase 2: sim = sigmoid @ w_p[s]^T (WMMA, N=16),
// max over k columns via shfl, store z = max*mask.
// Dynamic LDS: [0,64K) sigmoid tiles, [64K,72K) staged w_p fragments.
// ---------------------------------------------------------------------------
__global__ __launch_bounds__(256) void can_main_kernel(const float* __restrict__ x,
                                                       const float* __restrict__ mask,
                                                       const unsigned* __restrict__ wtp,
                                                       const unsigned* __restrict__ wpp,
                                                       float* __restrict__ simz,
                                                       unsigned short* __restrict__ se_out) {
    extern __shared__ char dyn_smem[];
    unsigned short* smem = (unsigned short*)dyn_smem;          // 64 KB sigmoid tiles
    const char* wpstage  = dyn_smem + 65536;                   // 8 KB TDM-staged w_p

    const int tid   = threadIdx.x;
    const int wave  = tid >> 5;
    const int lane  = tid & 31;
    const int lhalf = (lane & 16) ? 1 : 0;
    const int ln    = lane & 15;
    const int s     = blockIdx.y;
    const int rowbase = blockIdx.x * 128 + wave * 16;

    // --- TDM: async-load this s's 8 KB of w_p fragments into LDS @ 65536.
    // D# per CDNA5 ISA: 1-D copy, data_size=8B (code 3),
    // tensor_dim0 = tile_dim0 = 1024 units, type=2, count=1.
    if (wave == 0) {
        unsigned long long ga =
            (unsigned long long)(uintptr_t)(wpp + (size_t)s * 2048);
        v4u g0 = { 1u,                                   // count=1 (valid D#)
                   65536u,                               // lds_addr
                   (unsigned)ga,                         // global_addr[31:0]
                   ((unsigned)(ga >> 32) & 0x1FFFFFFu) | 0x80000000u }; // [56:32] | type=2
        v8i g1 = { (int)0x30000u,          // data_size=3 (8B units)
                   (int)(1024u << 16),     // tensor_dim0[15:0] << 16
                   (int)(1u << 16),        // tensor_dim1 = 1
                   (int)(1024u << 16),     // tile_dim0 = 1024
                   0,                      // tile_dim1 = tile_dim2 = 0
                   1024,                   // tensor_dim0_stride = 1024
                   0, 0 };
        v4i gz4 = { 0, 0, 0, 0 };
        v8i gz8 = { 0, 0, 0, 0, 0, 0, 0, 0 };
        __builtin_amdgcn_tensor_load_to_lds(g0, g1, gz4, gz4, gz8, 0);
    }

    // per-lane masks for the 8 C rows it carries (rows r / r+8 by lane half)
    float mrow[8];
#pragma unroll
    for (int r = 0; r < 8; ++r)
        mrow[r] = mask[(size_t)s * B_DIM + rowbase + r + (lhalf ? 8 : 0)];

    // --- Load A fragments (16 rows x 256 K) straight from global, pack bf16.
    // ISA 16-bit A 16x32 layout: lanes 0-15 K runs {0..7,16..23}; lanes 16-31
    // K runs {8..15,24..31}; row M = lane&15.
    Frag a[8];
    {
        const float* xr = x + ((size_t)s * B_DIM + rowbase + ln) * D_DIM;
#pragma unroll
        for (int kk = 0; kk < 8; ++kk) {
            const int base = kk * 32 + (lhalf ? 8 : 0);
            float4 r0 = *(const float4*)(xr + base);
            float4 r1 = *(const float4*)(xr + base + 4);
            float4 r2 = *(const float4*)(xr + base + 16);
            float4 r3 = *(const float4*)(xr + base + 20);
            a[kk].u[0] = bf16_rne(r0.x) | (bf16_rne(r0.y) << 16);
            a[kk].u[1] = bf16_rne(r0.z) | (bf16_rne(r0.w) << 16);
            a[kk].u[2] = bf16_rne(r1.x) | (bf16_rne(r1.y) << 16);
            a[kk].u[3] = bf16_rne(r1.z) | (bf16_rne(r1.w) << 16);
            a[kk].u[4] = bf16_rne(r2.x) | (bf16_rne(r2.y) << 16);
            a[kk].u[5] = bf16_rne(r2.z) | (bf16_rne(r2.w) << 16);
            a[kk].u[6] = bf16_rne(r3.x) | (bf16_rne(r3.y) << 16);
            a[kk].u[7] = bf16_rne(r3.z) | (bf16_rne(r3.w) << 16);
        }
    }

    // warm L2 for the first w_t fragment batch
    __builtin_prefetch(wtp + (size_t)lane * 8, 0, 1);

    // --- Phase 1: 16 output tiles of 16 cols each across O=256
    for (int nt = 0; nt < 16; ++nt) {
        v8f acc = {};
#pragma unroll
        for (int kk = 0; kk < 8; ++kk) {
            Frag b;
            b.u8 = *(const v8u*)(wtp + ((size_t)((kk * 16 + nt) * 32 + lane)) * 8);
            acc = __builtin_amdgcn_wmma_f32_16x16x32_bf16(
                false, a[kk].v, false, b.v, (short)0, acc, false, false);
        }
        const int o = nt * 16 + ln;
#pragma unroll
        for (int r = 0; r < 8; ++r) {
            const int grow = rowbase + r + (lhalf ? 8 : 0);
            float se = acc[r] * mrow[r];
            se_out[((size_t)s * B_DIM + grow) * O_DIM + o] = (unsigned short)bf16_rne(se);
            float t = 1.0f / (1.0f + __expf(-se));
            smem[(wave * 16 + r + (lhalf ? 8 : 0)) * 256 + o] = (unsigned short)bf16_rne(t);
        }
    }

    // TDM data must be in LDS before any wave reads it in phase 2.
    if (wave == 0) __builtin_amdgcn_s_wait_tensorcnt(0);
    __syncthreads();

    // --- Phase 2: sim tile [16 rows x 16 k-cols] = sigmoid_strip @ w_p[s]^T
    v8f acc2 = {};
#pragma unroll
    for (int kk = 0; kk < 8; ++kk) {
        Frag a2, b2;
        const unsigned short* tp = &smem[(wave * 16 + ln) * 256 + kk * 32 + (lhalf ? 8 : 0)];
        a2.u4[0] = *(const v4u*)(tp);        // K run +0..7   (16B)
        a2.u4[1] = *(const v4u*)(tp + 16);   // K run +16..23 (16B)
        b2.u8 = *(const v8u*)(wpstage + (size_t)(kk * 32 + lane) * 32);
        acc2 = __builtin_amdgcn_wmma_f32_16x16x32_bf16(
            false, a2.v, false, b2.v, (short)0, acc2, false, false);
    }

    // max over k (valid cols 0..9), cross-lane within each 16-lane half
#pragma unroll
    for (int r = 0; r < 8; ++r) {
        float v = (ln < K_DIM) ? acc2[r] : -__builtin_inff();
#pragma unroll
        for (int off = 1; off < 16; off <<= 1)
            v = fmaxf(v, __shfl_xor(v, off, 32));
        if (ln == 0) {
            const int grow = rowbase + r + (lhalf ? 8 : 0);
            simz[(size_t)s * B_DIM + grow] = v * mrow[r];   // z = max_k(sim) * mask
        }
    }
}

// ---------------------------------------------------------------------------
// Finalize: per block = one b.  alpha = mask*exp(z-max)/sum(mask*exp(z-max));
// paa_h[b,o] = sum_s se[s,b,o]*alpha[s].
// Threads split: o-pair = tid&127 (dword bf16x2 loads), S halved by tid>>7.
// ---------------------------------------------------------------------------
__global__ __launch_bounds__(256) void can_finalize_kernel(const float* __restrict__ simz,
                                                           const float* __restrict__ mask,
                                                           const unsigned short* __restrict__ se,
                                                           float* __restrict__ out) {
    __shared__ float alpha[S_DIM];
    __shared__ float part[O_DIM];
    __shared__ float redm[8];
    __shared__ float reds[8];

    const int b    = blockIdx.x;
    const int tid  = threadIdx.x;
    const int wave = tid >> 5;
    const int lane = tid & 31;

    float z0 = simz[(size_t)tid * B_DIM + b];
    float z1 = simz[(size_t)(tid + 256) * B_DIM + b];
    float m0 = mask[(size_t)tid * B_DIM + b];
    float m1 = mask[(size_t)(tid + 256) * B_DIM + b];

    // global max of z over S
    float v = fmaxf(z0, z1);
#pragma unroll
    for (int off = 16; off > 0; off >>= 1) v = fmaxf(v, __shfl_xor(v, off, 32));
    if (lane == 0) redm[wave] = v;
    __syncthreads();
    float gmax = redm[0];
#pragma unroll
    for (int i = 1; i < 8; ++i) gmax = fmaxf(gmax, redm[i]);

    float e0 = m0 * __expf(z0 - gmax);
    float e1 = m1 * __expf(z1 - gmax);
    float sv = e0 + e1;
#pragma unroll
    for (int off = 16; off > 0; off >>= 1) sv += __shfl_xor(sv, off, 32);
    if (lane == 0) reds[wave] = sv;
    __syncthreads();
    float gsum = reds[0];
#pragma unroll
    for (int i = 1; i < 8; ++i) gsum += reds[i];
    float inv = 1.0f / gsum;

    alpha[tid]       = e0 * inv;
    alpha[tid + 256] = e1 * inv;
    __syncthreads();

    // weighted reduction over S; dword loads of bf16 o-pairs, S split in half
    const int p    = tid & 127;       // o-pair index (o = 2p, 2p+1)
    const int half = tid >> 7;        // S half
    const unsigned* pw = (const unsigned*)(se + (size_t)b * O_DIM) + p;
    float a0 = 0.0f, a1 = 0.0f;
    const int s0 = half * 256;
    for (int ss = s0; ss < s0 + 256; ++ss) {
        unsigned w = pw[(size_t)ss * (B_DIM * O_DIM / 2)];
        float al = alpha[ss];
        a0 = fmaf(al, bf16_to_f((unsigned short)(w & 0xffffu)), a0);
        a1 = fmaf(al, bf16_to_f((unsigned short)(w >> 16)), a1);
    }
    if (half == 0) {
        part[2 * p]     = a0;
        part[2 * p + 1] = a1;
    }
    __syncthreads();
    if (half == 1) {
        out[(size_t)b * O_DIM + 2 * p]     = part[2 * p] + a0;
        out[(size_t)b * O_DIM + 2 * p + 1] = part[2 * p + 1] + a1;
    }
}

// ---------------------------------------------------------------------------
extern "C" void kernel_launch(void* const* d_in, const int* in_sizes, int n_in,
                              void* d_out, int out_size, void* d_ws, size_t ws_size,
                              hipStream_t stream) {
    const float* x    = (const float*)d_in[0];   // [S,B,D]
    const float* mask = (const float*)d_in[1];   // [S,B]
    const float* w_t  = (const float*)d_in[2];   // [D,O]
    const float* w_p  = (const float*)d_in[3];   // [K,S,O]
    float* out = (float*)d_out;                  // [B,O]

    char* ws = (char*)d_ws;
    unsigned* wtp        = (unsigned*)(ws);                                   // 128 KB
    unsigned* wpp        = (unsigned*)(ws + 131072);                          // 4 MB
    float* simz          = (float*)(ws + 131072 + 4194304);                   // 2 MB
    unsigned short* sebf = (unsigned short*)(ws + 131072 + 4194304 + 2097152);// 256 MB

    pack_wt_kernel<<<128, 256, 0, stream>>>(w_t, wtp);
    pack_wp_kernel<<<4096, 256, 0, stream>>>(w_p, wpp);
    can_main_kernel<<<dim3(B_DIM / 128, S_DIM), 256, 73728, stream>>>(
        x, mask, wtp, wpp, simz, sebf);
    can_finalize_kernel<<<B_DIM, 256, 0, stream>>>(simz, mask, sebf, out);
}
